// StableDynamicsHybridRecurrentFFN_69123203662007
// MI455X (gfx1250) — compile-verified, matrix operand order checked by reading, not run on hardware
//
#include <hip/hip_runtime.h>

// ---- problem constants --------------------------------------------------
#define B_    8
#define S_    1024
#define H_    2048
#define REC_  2048
#define HALF_ 1024
#define M_    (B_ * S_)     // 8192 rows for every GEMM

typedef __attribute__((ext_vector_type(16))) __bf16 v16bf;
typedef __attribute__((ext_vector_type(8)))  __bf16 v8bf;
typedef __attribute__((ext_vector_type(4)))  __bf16 v4bf;
typedef __attribute__((ext_vector_type(8)))  float  v8f;

__device__ __forceinline__ __bf16 f2bf(float f) {
    unsigned u = __builtin_bit_cast(unsigned, f);
    u += 0x7FFFu + ((u >> 16) & 1u);              // round-to-nearest-even
    unsigned short h = (unsigned short)(u >> 16);
    return __builtin_bit_cast(__bf16, h);
}

// ---- CDNA5 async-copy helpers (gfx1250 ASYNCcnt path) -------------------
// LDS byte offset of a __shared__ object: generic address = {aperture, off}
__device__ __forceinline__ unsigned lds_off_u32(const void* p) {
    return (unsigned)(unsigned long long)p;
}

// global -> LDS direct copy, 16 bytes per lane, tracked by ASYNCcnt
__device__ __forceinline__ void async_copy_b128(unsigned lds_off, const void* gaddr) {
    asm volatile("global_load_async_to_lds_b128 %0, %1, off"
                 :: "v"(lds_off), "v"(gaddr)
                 : "memory");
}

template <int N>
__device__ __forceinline__ void wait_asynccnt() {
    asm volatile("s_wait_asynccnt %0" :: "i"(N) : "memory");
}

// ---- fp32 -> bf16 conversion -------------------------------------------
__global__ __launch_bounds__(256) void cvt_f32_to_bf16(
    const float* __restrict__ in, __bf16* __restrict__ out, int n4) {
    int i = blockIdx.x * blockDim.x + threadIdx.x;
    if (i < n4) {
        float4 v = ((const float4*)in)[i];
        v4bf r = { f2bf(v.x), f2bf(v.y), f2bf(v.z), f2bf(v.w) };
        ((v4bf*)out)[i] = r;
    }
}

// ---- hidden = up * silu(gate)  (silu already applied to g) -------------
__global__ __launch_bounds__(256) void hidden_mul_kernel(
    const float* __restrict__ u, const float* __restrict__ g,
    __bf16* __restrict__ h, int n4) {
    int i = blockIdx.x * blockDim.x + threadIdx.x;
    if (i < n4) {
        float4 a = ((const float4*)u)[i];
        float4 b = ((const float4*)g)[i];
        v4bf r = { f2bf(a.x * b.x), f2bf(a.y * b.y),
                   f2bf(a.z * b.z), f2bf(a.w * b.w) };
        ((v4bf*)h)[i] = r;
    }
}

// ---- epilogue activations ----------------------------------------------
// 0: identity  1: pi*tanh  2: exp(-softplus(v)) == sigmoid(-v)
// 3: sigmoid   4: silu
template <int ACT>
__device__ __forceinline__ float apply_act(float v) {
    if constexpr (ACT == 1) return 3.14159274101257f * tanhf(v);
    else if constexpr (ACT == 2) return 1.0f / (1.0f + __expf(v));
    else if constexpr (ACT == 3) return 1.0f / (1.0f + __expf(-v));
    else if constexpr (ACT == 4) return v / (1.0f + __expf(-v));
    else return v;
}

// ---- bf16 WMMA GEMM:  C[M x N] (+)= act(A[M x K] * W[N x K]^T) ---------
// Block tile 128x128, K-step 32, 8 waves (wave32), each wave: 2x4 frags.
// Double-buffered LDS fed by global_load_async_to_lds_b128 (2 tiles in
// flight = 8 async ops/wave; s_wait_asynccnt 4 retires the oldest tile).
template <int ACT, int ACCUM>
__global__ __launch_bounds__(256) void gemm_bf16_wmma(
    const __bf16* __restrict__ A, const __bf16* __restrict__ W,
    float* __restrict__ C, int N, int K) {
    __shared__ __bf16 As[2][128][40];   // +8 pad: 80B row stride, 16B aligned
    __shared__ __bf16 Bs[2][128][40];

    const int tid  = threadIdx.x;
    const int lane = tid & 31;
    const int wave = tid >> 5;
    const int wm   = wave >> 1;          // 0..3 -> M sub-tile
    const int wn   = wave & 1;           // 0..1 -> N sub-tile
    const int bm   = blockIdx.y * 128;
    const int bn   = blockIdx.x * 128;

    const v8f zero = {0.f, 0.f, 0.f, 0.f, 0.f, 0.f, 0.f, 0.f};
    v8f acc[2][4];
#pragma unroll
    for (int mi = 0; mi < 2; ++mi)
#pragma unroll
        for (int ni = 0; ni < 4; ++ni) acc[mi][ni] = zero;

    // this thread's two staging chunks (8 bf16 = 16B each) per tile
    const int r0 = tid >> 2,          c0 = (tid & 3) * 8;          // ids 0..255
    const int r1 = (tid + 256) >> 2,  c1 = ((tid + 256) & 3) * 8;  // ids 256..511
    const __bf16* gA0 = A + (size_t)(bm + r0) * K + c0;
    const __bf16* gA1 = A + (size_t)(bm + r1) * K + c1;
    const __bf16* gB0 = W + (size_t)(bn + r0) * K + c0;
    const __bf16* gB1 = W + (size_t)(bn + r1) * K + c1;
    unsigned lA0[2], lA1[2], lB0[2], lB1[2];
#pragma unroll
    for (int b = 0; b < 2; ++b) {
        lA0[b] = lds_off_u32(&As[b][r0][c0]);
        lA1[b] = lds_off_u32(&As[b][r1][c1]);
        lB0[b] = lds_off_u32(&Bs[b][r0][c0]);
        lB1[b] = lds_off_u32(&Bs[b][r1][c1]);
    }

    auto issue_tile = [&](int buf, int k0) {
        async_copy_b128(lA0[buf], gA0 + k0);
        async_copy_b128(lA1[buf], gA1 + k0);
        async_copy_b128(lB0[buf], gB0 + k0);
        async_copy_b128(lB1[buf], gB1 + k0);
    };

    // per-lane WMMA fragment addressing (16-bit A/B layout, ISA 7.12.2):
    // lane L: row = L&15, K-chunks at (L>>4)*8 and (L>>4)*8 + 16
    const int frow = lane & 15;
    const int fk   = (lane >> 4) * 8;

    const int nk = K >> 5;              // K / 32
    issue_tile(0, 0);
    issue_tile(1, 32);

    for (int k = 0; k < nk; ++k) {
        const int cur = k & 1;
        if (k + 1 < nk) wait_asynccnt<4>();   // oldest tile (k) landed
        else            wait_asynccnt<0>();   // tail: nothing else in flight
        __syncthreads();                      // tile k visible block-wide

        v16bf afrag[2], bfrag[4];
#pragma unroll
        for (int mi = 0; mi < 2; ++mi) {
            int row = wm * 32 + mi * 16 + frow;
            v8bf lo = *(const v8bf*)&As[cur][row][fk];
            v8bf hi = *(const v8bf*)&As[cur][row][fk + 16];
            afrag[mi] = __builtin_shufflevector(lo, hi, 0, 1, 2, 3, 4, 5, 6, 7,
                                                8, 9, 10, 11, 12, 13, 14, 15);
        }
#pragma unroll
        for (int ni = 0; ni < 4; ++ni) {
            int row = wn * 64 + ni * 16 + frow;
            v8bf lo = *(const v8bf*)&Bs[cur][row][fk];
            v8bf hi = *(const v8bf*)&Bs[cur][row][fk + 16];
            bfrag[ni] = __builtin_shufflevector(lo, hi, 0, 1, 2, 3, 4, 5, 6, 7,
                                                8, 9, 10, 11, 12, 13, 14, 15);
        }
#pragma unroll
        for (int mi = 0; mi < 2; ++mi)
#pragma unroll
            for (int ni = 0; ni < 4; ++ni)
                acc[mi][ni] = __builtin_amdgcn_wmma_f32_16x16x32_bf16(
                    false, afrag[mi], false, bfrag[ni],
                    (short)0, acc[mi][ni], false, false);

        __syncthreads();                      // everyone done reading buf[cur]
        if (k + 2 < nk) issue_tile(cur, (k + 2) * 32);
    }

    // C layout (ISA 7.12.2): vgpr v, lanes0-15: M=v, N=lane; lanes16-31: M=8+v
    const int mofs = (lane >> 4) * 8;
    const int nofs = lane & 15;
#pragma unroll
    for (int mi = 0; mi < 2; ++mi) {
#pragma unroll
        for (int ni = 0; ni < 4; ++ni) {
            int row0 = bm + wm * 32 + mi * 16 + mofs;
            int col  = bn + wn * 64 + ni * 16 + nofs;
#pragma unroll
            for (int v = 0; v < 8; ++v) {
                float r = apply_act<ACT>(acc[mi][ni][v]);
                float* p = C + (size_t)(row0 + v) * N + col;
                if constexpr (ACCUM) *p += r; else *p = r;
            }
        }
    }
}

// ---- recurrence: 8192 independent 2D-rotation chains over S steps ------
__global__ __launch_bounds__(256) void recurrence_kernel(
    const float* __restrict__ omega, const float* __restrict__ damp,
    const float* __restrict__ gate,  const float* __restrict__ cand,
    const float* __restrict__ q,     const float* __restrict__ init,
    __bf16* __restrict__ readouts) {
    int idx = blockIdx.x * blockDim.x + threadIdx.x;   // 0..8191
    int b = idx >> 10;          // / HALF_
    int p = idx & (HALF_ - 1);
    float se = init[p];
    float so = init[p + HALF_];
    size_t rowO = (size_t)b * S_ * HALF_ + p;
    size_t rowR = (size_t)b * S_ * REC_ + p;
    for (int s = 0; s < S_; ++s) {
        float om = omega[rowO];
        float sn, cs;
        __sincosf(om, &sn, &cs);
        float dpe = damp[rowR], dpo = damp[rowR + HALF_];
        float gte = gate[rowR], gto = gate[rowR + HALF_];
        float cde = cand[rowR], cdo = cand[rowR + HALF_];
        float qe  = q[rowR],    qo  = q[rowR + HALF_];
        float re = cs * se - sn * so;
        float ro = sn * se + cs * so;
        se = dpe * re + gte * cde;
        so = dpo * ro + gto * cdo;
        float ve = qe * se, vo = qo * so;
        readouts[rowR]         = f2bf(ve / (1.0f + __expf(-ve)));
        readouts[rowR + HALF_] = f2bf(vo / (1.0f + __expf(-vo)));
        rowO += HALF_;
        rowR += REC_;
    }
}

// ---- host orchestration -------------------------------------------------
extern "C" void kernel_launch(void* const* d_in, const int* in_sizes, int n_in,
                              void* d_out, int out_size, void* d_ws, size_t ws_size,
                              hipStream_t stream) {
    (void)in_sizes; (void)n_in; (void)out_size; (void)ws_size;
    const float* x    = (const float*)d_in[0];
    const float* Ww   = (const float*)d_in[1];
    const float* Wg   = (const float*)d_in[2];
    const float* Wo   = (const float*)d_in[3];
    const float* Wd   = (const float*)d_in[4];
    const float* Wq   = (const float*)d_in[5];
    const float* Wout = (const float*)d_in[6];
    const float* init = (const float*)d_in[7];
    const float* upW  = (const float*)d_in[8];
    const float* gtW  = (const float*)d_in[9];
    const float* dnW  = (const float*)d_in[10];
    float* out = (float*)d_out;

    char*  ws  = (char*)d_ws;
    size_t off = 0;
    auto alloc = [&](size_t bytes) -> char* {
        char* p = ws + off;
        off = (off + bytes + 255) & ~(size_t)255;
        return p;
    };

    const size_t MH = (size_t)M_ * H_;        // 16.7M
    __bf16* xb    = (__bf16*)alloc(MH * 2);
    __bf16* Wwb   = (__bf16*)alloc((size_t)REC_ * H_ * 2);
    __bf16* Wgb   = (__bf16*)alloc((size_t)REC_ * H_ * 2);
    __bf16* Wob   = (__bf16*)alloc((size_t)HALF_ * H_ * 2);
    __bf16* Wdb   = (__bf16*)alloc((size_t)REC_ * H_ * 2);
    __bf16* Wqb   = (__bf16*)alloc((size_t)REC_ * H_ * 2);
    __bf16* Woutb = (__bf16*)alloc((size_t)H_ * REC_ * 2);
    __bf16* upb   = (__bf16*)alloc((size_t)REC_ * H_ * 2);
    __bf16* gtb   = (__bf16*)alloc((size_t)REC_ * H_ * 2);
    __bf16* dnb   = (__bf16*)alloc((size_t)H_ * REC_ * 2);
    float*  omg   = (float*)alloc((size_t)M_ * HALF_ * 4);
    float*  bufA  = (float*)alloc((size_t)M_ * REC_ * 4);  // up result, later damp
    float*  bufB  = (float*)alloc((size_t)M_ * REC_ * 4);  // silu(gateW), later gate
    float*  candb = (float*)alloc((size_t)M_ * REC_ * 4);
    float*  qb    = (float*)alloc((size_t)M_ * REC_ * 4);
    __bf16* hid   = (__bf16*)alloc((size_t)M_ * REC_ * 2);
    __bf16* rdo   = (__bf16*)alloc((size_t)M_ * REC_ * 2);

    auto cvt = [&](const float* src, __bf16* dst, size_t n) {
        int n4 = (int)(n / 4);
        cvt_f32_to_bf16<<<(n4 + 255) / 256, 256, 0, stream>>>(src, dst, n4);
    };
    cvt(x, xb, MH);
    cvt(Ww, Wwb, (size_t)REC_ * H_);
    cvt(Wg, Wgb, (size_t)REC_ * H_);
    cvt(Wo, Wob, (size_t)HALF_ * H_);
    cvt(Wd, Wdb, (size_t)REC_ * H_);
    cvt(Wq, Wqb, (size_t)REC_ * H_);
    cvt(Wout, Woutb, (size_t)H_ * REC_);
    cvt(upW, upb, (size_t)REC_ * H_);
    cvt(gtW, gtb, (size_t)REC_ * H_);
    cvt(dnW, dnb, (size_t)H_ * REC_);

    dim3 blk(256);
    dim3 gN2(REC_ / 128, M_ / 128);
    dim3 gN1(HALF_ / 128, M_ / 128);
    dim3 gOut(H_ / 128, M_ / 128);

    // FFN branch first (bufA/bufB are reused by the projection GEMMs after)
    gemm_bf16_wmma<0, 0><<<gN2, blk, 0, stream>>>(xb, upb, bufA, REC_, H_);
    gemm_bf16_wmma<4, 0><<<gN2, blk, 0, stream>>>(xb, gtb, bufB, REC_, H_);
    {
        int n4 = (int)((size_t)M_ * REC_ / 4);
        hidden_mul_kernel<<<(n4 + 255) / 256, 256, 0, stream>>>(bufA, bufB, hid, n4);
    }

    // projections
    gemm_bf16_wmma<1, 0><<<gN1, blk, 0, stream>>>(xb, Wob, omg, HALF_, H_);
    gemm_bf16_wmma<2, 0><<<gN2, blk, 0, stream>>>(xb, Wdb, bufA, REC_, H_);
    gemm_bf16_wmma<3, 0><<<gN2, blk, 0, stream>>>(xb, Wgb, bufB, REC_, H_);
    gemm_bf16_wmma<4, 0><<<gN2, blk, 0, stream>>>(xb, Wwb, candb, REC_, H_);
    gemm_bf16_wmma<0, 0><<<gN2, blk, 0, stream>>>(xb, Wqb, qb, REC_, H_);

    // sequential scan, parallel across B*HALF channels
    recurrence_kernel<<<(B_ * HALF_) / 256, 256, 0, stream>>>(
        omg, bufA, bufB, candb, qb, init, rdo);

    // output GEMMs: store then accumulate
    gemm_bf16_wmma<0, 0><<<gOut, blk, 0, stream>>>(rdo, Woutb, out, H_, REC_);
    gemm_bf16_wmma<0, 1><<<gOut, blk, 0, stream>>>(hid, dnb, out, H_, REC_);
}
